// BandMHA_22196390986583
// MI455X (gfx1250) — matrix operand order repaired and checked by reading, not compiled, not run on hardware
//
#include <hip/hip_runtime.h>
#include <hip/hip_bf16.h>
#include <math.h>

// ---------------------------------------------------------------------------
// Banded MHA (B=2, L=1024, D=1024, H=16, DH=64, W=64) for gfx1250 (CDNA5).
// All matmuls via v_wmma_f32_16x16x32_f16 (wave32 WMMA), fp32 accumulate.
// 5 launches: cvt_all -> pad_kv -> fused QKV GEMM -> windowed attention -> out GEMM
// ---------------------------------------------------------------------------

typedef __attribute__((ext_vector_type(16))) _Float16 v16h;
typedef __attribute__((ext_vector_type(8)))  _Float16 v8h;
typedef __attribute__((ext_vector_type(8)))  float    v8f;

static constexpr int Bsz   = 2;
static constexpr int Lseq  = 1024;
static constexpr int Dm    = 1024;
static constexpr int Hh    = 16;
static constexpr int DHd   = 64;
static constexpr int Wwin  = 64;
static constexpr int MROWS = Bsz * Lseq;        // 2048
static constexpr int KROWS = Lseq + Wwin;       // 1088: 63 pad + 1024 + 1 guard row
static constexpr int VLEN  = 1120;              // padded key axis for v_t (mult. of 16)

#define WMMA16(a, b, c) \
  __builtin_amdgcn_wmma_f32_16x16x32_f16(false, (a), false, (b), (short)0, (c), false, false)

// ---------------------------------------------------------------------------
// Fused fp32 -> f16 conversion for x + 4 weight matrices, 8 elements/thread.
// ---------------------------------------------------------------------------
__global__ void cvt_all_kernel(const float* __restrict__ x,
                               const float* __restrict__ wq,
                               const float* __restrict__ wk,
                               const float* __restrict__ wv,
                               const float* __restrict__ wo,
                               _Float16* __restrict__ x_h,
                               _Float16* __restrict__ wq_h,
                               _Float16* __restrict__ wk_h,
                               _Float16* __restrict__ wv_h,
                               _Float16* __restrict__ wo_h) {
  const size_t NX = (size_t)MROWS * Dm;     // 2,097,152
  const size_t NW = (size_t)Dm * Dm;        // 1,048,576
  size_t e = ((size_t)blockIdx.x * blockDim.x + threadIdx.x) * 8;
  if (e >= NX + 4 * NW) return;

  const float* src;
  _Float16*    dst;
  size_t       off;
  if (e < NX) {
    src = x; dst = x_h; off = e;
  } else {
    size_t r = e - NX;
    int wsel = (int)(r / NW);
    off = r % NW;
    src = (wsel == 0) ? wq   : (wsel == 1) ? wk   : (wsel == 2) ? wv   : wo;
    dst = (wsel == 0) ? wq_h : (wsel == 1) ? wk_h : (wsel == 2) ? wv_h : wo_h;
  }
  float4 a = *(const float4*)(src + off);
  float4 b = *(const float4*)(src + off + 4);
  v8h o;
  o[0] = (_Float16)a.x; o[1] = (_Float16)a.y; o[2] = (_Float16)a.z; o[3] = (_Float16)a.w;
  o[4] = (_Float16)b.x; o[5] = (_Float16)b.y; o[6] = (_Float16)b.z; o[7] = (_Float16)b.w;
  *(v8h*)(dst + off) = o;
}

// ---------------------------------------------------------------------------
// Window-history padding: last_k_init -> k_h[:, 0:63, :], last_v_init -> v_t[:, :, :, 0:63]
// ---------------------------------------------------------------------------
__global__ void pad_kv_kernel(const float* __restrict__ lastk,
                              const float* __restrict__ lastv,
                              _Float16* __restrict__ k_h,
                              _Float16* __restrict__ v_t) {
  const int npad = Bsz * 63 * Dm;           // 129,024 per tensor
  int i = blockIdx.x * blockDim.x + threadIdx.x;
  if (i >= 2 * npad) return;
  bool is_v = (i >= npad);
  int j   = is_v ? (i - npad) : i;
  int b   = j / (63 * Dm);
  int rem = j % (63 * Dm);
  int p   = rem / Dm;                       // 0..62
  int f   = rem % Dm;                       // h*64 + dh
  if (!is_v) {
    k_h[(size_t)(b * KROWS + p) * Dm + f] = (_Float16)lastk[rem];
  } else {
    int h = f >> 6, dh = f & 63;
    v_t[((size_t)((b * Hh + h) * DHd + dh)) * VLEN + p] = (_Float16)lastv[rem];
  }
}

// ---------------------------------------------------------------------------
// Shared WMMA GEMM core: C[m,n] = sum_d A[m,d] * W[n,d]
// One wave -> 64x64 C tile, 4x4 f32 accumulators, K in steps of 32.
// ---------------------------------------------------------------------------
__device__ __forceinline__ void gemm_accum(const _Float16* __restrict__ A,
                                           const _Float16* __restrict__ Wt,
                                           int m0, int n0, int hs, int l16,
                                           v8f acc[4][4]) {
#pragma unroll
  for (int i = 0; i < 4; ++i)
#pragma unroll
    for (int j = 0; j < 4; ++j)
#pragma unroll
      for (int r = 0; r < 8; ++r) acc[i][j][r] = 0.0f;

  for (int ks = 0; ks < Dm / 32; ++ks) {
    // A fragment (16x32): lane holds row m0+i*16+l16; two 8-half chunks at
    // K = ks*32 + hs*16 and +8 (contiguous, matches B).
    v16h afr[4];
    int abase = ks * 32 + hs * 16;
#pragma unroll
    for (int i = 0; i < 4; ++i) {
      const _Float16* p = A + (size_t)(m0 + i * 16 + l16) * Dm + abase;
      union { v16h v; v8h h[2]; } u;
      u.h[0] = *(const v8h*)p;
      u.h[1] = *(const v8h*)(p + 8);
      afr[i] = u.v;
    }
    // B fragment (32x16 K x N): lane holds col n0+j*16+l16 = row of W;
    // 16 contiguous halves at K = ks*32 + hs*16.
    v16h bfr[4];
    int bbase = ks * 32 + hs * 16;
#pragma unroll
    for (int j = 0; j < 4; ++j)
      bfr[j] = *(const v16h*)(Wt + (size_t)(n0 + j * 16 + l16) * Dm + bbase);

#pragma unroll
    for (int i = 0; i < 4; ++i)
#pragma unroll
      for (int j = 0; j < 4; ++j)
        acc[i][j] = WMMA16(afr[i], bfr[j], acc[i][j]);
  }
}

// Writeback. C layout per lane: VGPR r -> M = r + 8*hs, N = l16.
// MODE 0: q_h row-major f16       MODE 1: k_h (b*KROWS + 63 + l) f16
// MODE 2: v_t transposed f16      MODE 3: fp32 row-major
template <int MODE>
__device__ __forceinline__ void gemm_store(_Float16* __restrict__ Ch,
                                           float* __restrict__ Cf,
                                           int m0, int n0, int hs, int l16,
                                           v8f acc[4][4]) {
#pragma unroll
  for (int i = 0; i < 4; ++i) {
#pragma unroll
    for (int j = 0; j < 4; ++j) {
#pragma unroll
      for (int r = 0; r < 8; ++r) {
        int m = m0 + i * 16 + hs * 8 + r;
        int n = n0 + j * 16 + l16;
        float val = acc[i][j][r];
        if (MODE == 0) {
          Ch[(size_t)m * Dm + n] = (_Float16)val;
        } else if (MODE == 1) {
          int b = m >> 10, l = m & 1023;
          Ch[(size_t)(b * KROWS + 63 + l) * Dm + n] = (_Float16)val;
        } else if (MODE == 2) {
          int b = m >> 10, l = m & 1023;
          int h = n >> 6, dh = n & 63;
          Ch[((size_t)((b * Hh + h) * DHd + dh)) * VLEN + 63 + l] = (_Float16)val;
        } else {
          Cf[(size_t)m * Dm + n] = val;
        }
      }
    }
  }
}

// Fused QKV projection: blockIdx.y selects {q, k, v}. 1536 waves in flight.
__global__ __launch_bounds__(32) void gemm_qkv_kernel(
    const _Float16* __restrict__ x_h,
    const _Float16* __restrict__ wq, const _Float16* __restrict__ wk,
    const _Float16* __restrict__ wv,
    _Float16* __restrict__ q_h, _Float16* __restrict__ k_h,
    _Float16* __restrict__ v_t) {
  const int NT = Dm / 64;
  int m0 = (blockIdx.x / NT) * 64, n0 = (blockIdx.x % NT) * 64;
  int lane = threadIdx.x, hs = lane >> 4, l16 = lane & 15;
  int which = blockIdx.y;

  const _Float16* W = (which == 0) ? wq : (which == 1) ? wk : wv;
  v8f acc[4][4];
  gemm_accum(x_h, W, m0, n0, hs, l16, acc);
  if (which == 0)      gemm_store<0>(q_h, nullptr, m0, n0, hs, l16, acc);
  else if (which == 1) gemm_store<1>(k_h, nullptr, m0, n0, hs, l16, acc);
  else                 gemm_store<2>(v_t, nullptr, m0, n0, hs, l16, acc);
}

// Output projection: attn_h @ Wo.T -> fp32 out.
__global__ __launch_bounds__(32) void gemm_out_kernel(
    const _Float16* __restrict__ attn_h, const _Float16* __restrict__ wo,
    float* __restrict__ out) {
  const int NT = Dm / 64;
  int m0 = (blockIdx.x / NT) * 64, n0 = (blockIdx.x % NT) * 64;
  int lane = threadIdx.x, hs = lane >> 4, l16 = lane & 15;
  v8f acc[4][4];
  gemm_accum(attn_h, wo, m0, n0, hs, l16, acc);
  gemm_store<3>(nullptr, out, m0, n0, hs, l16, acc);
}

// ---------------------------------------------------------------------------
// Windowed attention: one wave per (b, h, 16-query block).
// scores (16 x 80, 5 key tiles) -> banded mask -> softmax (all 32 lanes) ->
// probs (16 x 96, zero-padded) x v (96 x 64) -> attn_h f16.
// ---------------------------------------------------------------------------
__global__ __launch_bounds__(32) void attn_wmma_kernel(
    const _Float16* __restrict__ q_h, const _Float16* __restrict__ k_h,
    const _Float16* __restrict__ v_t, _Float16* __restrict__ attn_h) {
  __shared__ __align__(64) float    sc[16][80];
  __shared__ __align__(64) _Float16 probs[16][96];
  __shared__ __align__(64) float    pmax[16][2];
  __shared__ __align__(64) float    psum[16][2];
  __shared__ __align__(64) float    inv_sum[16];

  int bid = blockIdx.x;            // b*H*64 + h*64 + lt
  int lt  = bid & 63;
  int h   = (bid >> 6) & (Hh - 1);
  int b   = bid >> 10;
  int l0  = lt * 16;

  int lane = threadIdx.x;
  int hs   = lane >> 4;
  int l16  = lane & 15;

  // q A-fragments (M=16 queries, K=64 features -> 2 k-steps)
  const _Float16* qp = q_h + (size_t)(b * Lseq + l0 + l16) * Dm + h * DHd;
  v16h qa[2];
#pragma unroll
  for (int ks = 0; ks < 2; ++ks) {
    union { v16h v; v8h h2[2]; } u;
    u.h2[0] = *(const v8h*)(qp + ks * 32 + hs * 16);
    u.h2[1] = *(const v8h*)(qp + ks * 32 + hs * 16 + 8);
    qa[ks] = u.v;
  }

  // Scores over 5 aligned key tiles covering padded keys [l0, l0+80)
#pragma unroll
  for (int t = 0; t < 5; ++t) {
    v8f acc;
#pragma unroll
    for (int r = 0; r < 8; ++r) acc[r] = 0.0f;
#pragma unroll
    for (int ks = 0; ks < 2; ++ks) {
      v16h kb = *(const v16h*)(k_h +
          (size_t)(b * KROWS + l0 + t * 16 + l16) * Dm + h * DHd +
          ks * 32 + hs * 16);
      acc = WMMA16(qa[ks], kb, acc);
    }
#pragma unroll
    for (int r = 0; r < 8; ++r) {
      int m = hs * 8 + r;          // query row within block
      int c = t * 16 + l16;        // padded-key offset from l0
      // valid window for query m: keys c in [m, m+63]
      sc[m][c] = (c >= m && c <= m + 63) ? acc[r] * 0.125f : -3.0e38f;
    }
  }
  __syncthreads();

  // Row softmax, all 32 lanes: lane handles row l16, columns [hs*40, hs*40+40).
  {
    int c0 = hs * 40;
    float mx = -3.0e38f;
    for (int c = 0; c < 40; ++c) mx = fmaxf(mx, sc[l16][c0 + c]);
    pmax[l16][hs] = mx;
    __syncthreads();
    mx = fmaxf(pmax[l16][0], pmax[l16][1]);
    float s = 0.0f;
    for (int c = 0; c < 40; ++c) {
      float e = __expf(sc[l16][c0 + c] - mx);
      s += e;
      probs[l16][c0 + c] = (_Float16)e;
    }
#pragma unroll
    for (int t = 0; t < 8; ++t) probs[l16][80 + hs * 8 + t] = (_Float16)0.0f;
    psum[l16][hs] = s;
    __syncthreads();
    if (lane < 16) inv_sum[lane] = 1.0f / (psum[lane][0] + psum[lane][1]);
    __syncthreads();
  }

  // AV: (16 x 96) x (96 x 64) -> 16 x 64, K padded to 96 (probs zero there).
  v8f oacc[4];
#pragma unroll
  for (int j = 0; j < 4; ++j)
#pragma unroll
    for (int r = 0; r < 8; ++r) oacc[j][r] = 0.0f;

#pragma unroll
  for (int ks = 0; ks < 3; ++ks) {
    union { v16h v; v8h h2[2]; } u;
    u.h2[0] = *(const v8h*)(&probs[l16][ks * 32 + hs * 16]);
    u.h2[1] = *(const v8h*)(&probs[l16][ks * 32 + hs * 16 + 8]);
    v16h pa = u.v;
#pragma unroll
    for (int j = 0; j < 4; ++j) {
      v16h vb = *(const v16h*)(v_t +
          (size_t)((b * Hh + h) * DHd + j * 16 + l16) * VLEN +
          l0 + ks * 32 + hs * 16);
      oacc[j] = WMMA16(pa, vb, oacc[j]);
    }
  }

#pragma unroll
  for (int j = 0; j < 4; ++j) {
#pragma unroll
    for (int r = 0; r < 8; ++r) {
      int m = hs * 8 + r;
      attn_h[(size_t)(b * Lseq + l0 + m) * Dm + h * DHd + j * 16 + l16] =
          (_Float16)(oacc[j][r] * inv_sum[m]);
    }
  }
}

// ---------------------------------------------------------------------------
// Host-side orchestration (5 launches total)
// ---------------------------------------------------------------------------
extern "C" void kernel_launch(void* const* d_in, const int* in_sizes, int n_in,
                              void* d_out, int out_size, void* d_ws, size_t ws_size,
                              hipStream_t stream) {
  (void)in_sizes; (void)n_in; (void)out_size; (void)ws_size;
  const float* x     = (const float*)d_in[0];
  const float* Wq    = (const float*)d_in[1];
  const float* Wk    = (const float*)d_in[2];
  const float* Wv    = (const float*)d_in[3];
  const float* Wo    = (const float*)d_in[4];
  const float* lastk = (const float*)d_in[5];
  const float* lastv = (const float*)d_in[6];
  float* out = (float*)d_out;

  // Workspace layout (all offsets 2 KB aligned)
  char* ws = (char*)d_ws;
  _Float16* x_h    = (_Float16*)(ws);                                   //  4 MiB
  _Float16* wq_h   = (_Float16*)(ws + (4u << 20));                      //  2 MiB each
  _Float16* wk_h   = (_Float16*)(ws + (6u << 20));
  _Float16* wv_h   = (_Float16*)(ws + (8u << 20));
  _Float16* wo_h   = (_Float16*)(ws + (10u << 20));
  _Float16* q_h    = (_Float16*)(ws + (12u << 20));                     //  4 MiB
  _Float16* k_h    = (_Float16*)(ws + (16u << 20));                     //  ~4.25 MiB
  _Float16* v_t    = (_Float16*)(ws + (21u << 20));                     //  ~4.4 MiB
  _Float16* attn_h = (_Float16*)(ws + (26u << 20));                     //  4 MiB

  const size_t ncvt = ((size_t)MROWS * Dm + 4 * (size_t)Dm * Dm) / 8;   // 786,432
  const int    npad2 = 2 * Bsz * 63 * Dm;                               // 258,048

  cvt_all_kernel<<<(unsigned)((ncvt + 255) / 256), 256, 0, stream>>>(
      x, Wq, Wk, Wv, Wo, x_h, wq_h, wk_h, wv_h, wo_h);
  pad_kv_kernel<<<(npad2 + 255) / 256, 256, 0, stream>>>(lastk, lastv, k_h, v_t);

  const int gemm_grid = (MROWS / 64) * (Dm / 64);                       // 512 tiles
  gemm_qkv_kernel<<<dim3(gemm_grid, 3), 32, 0, stream>>>(
      x_h, wq_h, wk_h, wv_h, q_h, k_h, v_t);

  attn_wmma_kernel<<<Bsz * Hh * (Lseq / 16), 32, 0, stream>>>(q_h, k_h, v_t, attn_h);

  gemm_out_kernel<<<gemm_grid, 32, 0, stream>>>(attn_h, wo_h, out);
}
